// IsoSymDisNet2D_82188494176367
// MI455X (gfx1250) — matrix operand order repaired and checked by reading, not compiled, not run on hardware
//
#include <hip/hip_runtime.h>
#include <hip/hip_bf16.h>

typedef __attribute__((ext_vector_type(16))) _Float16 v16h;
typedef __attribute__((ext_vector_type(8)))  _Float16 v8h;
typedef __attribute__((ext_vector_type(8)))  float    v8f;

#define PI_F 3.14159265358979323846f

// ---------------------------------------------------------------------------
// Zero-init the output accumulator (harness poisons d_out with 0xAA).
// ---------------------------------------------------------------------------
__global__ void zero_kernel(float* __restrict__ p, int n) {
    int i = blockIdx.x * 256 + threadIdx.x;
    if (i < n) p[i] = 0.0f;
}

// ---------------------------------------------------------------------------
// Band smoothing matrices: A_c[i,j] = (1/128) * sum_k m_c(k) cos(2*pi*k*(i-j)/128)
// m_c(k): |fftfreq| = min(k,128-k)/128 in [c/32,(c+1)/32] inclusive (exact
// integer test 4c <= min(k,128-k) <= 4c+4). A_c is symmetric (A^T == A).
// ---------------------------------------------------------------------------
__global__ void bandmat_kernel(_Float16* __restrict__ Ahalf) {
    int c = blockIdx.x;
    int lo = 4 * c, hi = 4 * c + 4;
    for (int idx = threadIdx.x; idx < 128 * 128; idx += 256) {
        int i = idx >> 7, j = idx & 127;
        int d = i - j;
        float acc = 0.0f;
        for (int k = 0; k < 128; ++k) {
            int f = (k < 64) ? k : (128 - k);
            if (f >= lo && f <= hi) {
                int m = (k * d) & 127;
                acc += __cosf((2.0f * PI_F / 128.0f) * (float)m);
            }
        }
        Ahalf[c * 16384 + idx] = (_Float16)(acc * (1.0f / 128.0f));
    }
}

// ---------------------------------------------------------------------------
// Convert each input image to f16 *transposed*: Xt[bv][x][y] = inp[bv][y][x].
// Makes GEMM1's B-fragments row-contiguous in LDS (2x ds_load_b128 per frag)
// and dedups the f32->f16 conversion across the 16 bands that reuse it.
// ---------------------------------------------------------------------------
__global__ void xprep_kernel(const float* __restrict__ inp,
                             _Float16* __restrict__ Xt) {
    int bv = blockIdx.x;                               // 64 images
    const float* src = inp + ((size_t)bv << 14);
    _Float16* dst = Xt + ((size_t)bv << 14);
    for (int idx = threadIdx.x; idx < 16384; idx += 256) {
        int y = idx >> 7, x = idx & 127;
        dst[x * 128 + y] = (_Float16)src[idx];
    }
}

// ---------------------------------------------------------------------------
// Effective conv weights with the lift weights folded through:
//   Weff[c,v,v',k,l] = sum_{n,g,a} lw[n,v]*lw[g,v']*rw[c,n,a]
//                      * Rot_{-2pi n/8}( kernels[c,a,(g-n) mod 8] )[k,l]
// ---------------------------------------------------------------------------
__global__ void weff_kernel(const float* __restrict__ kernels,
                            const float* __restrict__ rw,
                            float* __restrict__ weff) {
    int c = blockIdx.x;
    int t = threadIdx.x;
    if (t >= 49) return;
    int k = t / 7, l = t % 7;
    float gx = (2.0f * (float)l + 1.0f) / 7.0f - 1.0f;
    float gy = (2.0f * (float)k + 1.0f) / 7.0f - 1.0f;
    float acc00 = 0.f, acc01 = 0.f, acc10 = 0.f, acc11 = 0.f;
    for (int n = 0; n < 8; ++n) {
        float th = -2.0f * PI_F * (float)n / 8.0f;
        float cs = __cosf(th), sn = __sinf(th);
        float sx = cs * gx - sn * gy;
        float sy = sn * gx + cs * gy;
        float ix = ((sx + 1.0f) * 7.0f - 1.0f) * 0.5f;
        float iy = ((sy + 1.0f) * 7.0f - 1.0f) * 0.5f;
        float x0f = floorf(ix), y0f = floorf(iy);
        int   x0 = (int)x0f,    y0 = (int)y0f;
        float wx1 = ix - x0f, wx0 = 1.0f - wx1;
        float wy1 = iy - y0f, wy0 = 1.0f - wy1;
        float an = 2.0f * PI_F * (float)n / 8.0f;
        float lwn0 = __cosf(an), lwn1 = __sinf(an);
        for (int g = 0; g < 8; ++g) {
            int gs = (g - n) & 7;
            float ag = 2.0f * PI_F * (float)g / 8.0f;
            float lwg0 = __cosf(ag), lwg1 = __sinf(ag);
            float kv = 0.0f;
            for (int a = 0; a < 4; ++a) {
                const float* km = kernels + (((c * 4 + a) * 8 + gs) * 49);
                float s00 = ((unsigned)y0       < 7u && (unsigned)x0       < 7u) ? km[y0 * 7 + x0]           : 0.f;
                float s01 = ((unsigned)y0       < 7u && (unsigned)(x0 + 1) < 7u) ? km[y0 * 7 + x0 + 1]       : 0.f;
                float s10 = ((unsigned)(y0 + 1) < 7u && (unsigned)x0       < 7u) ? km[(y0 + 1) * 7 + x0]     : 0.f;
                float s11 = ((unsigned)(y0 + 1) < 7u && (unsigned)(x0 + 1) < 7u) ? km[(y0 + 1) * 7 + x0 + 1] : 0.f;
                float sv = s00 * (wy0 * wx0) + s01 * (wy0 * wx1)
                         + s10 * (wy1 * wx0) + s11 * (wy1 * wx1);
                kv = fmaf(rw[(c * 8 + n) * 4 + a], sv, kv);
            }
            acc00 = fmaf(lwn0 * lwg0, kv, acc00);
            acc01 = fmaf(lwn0 * lwg1, kv, acc01);
            acc10 = fmaf(lwn1 * lwg0, kv, acc10);
            acc11 = fmaf(lwn1 * lwg1, kv, acc11);
        }
    }
    weff[((c * 2 + 0) * 2 + 0) * 49 + t] = acc00;
    weff[((c * 2 + 0) * 2 + 1) * 49 + t] = acc01;
    weff[((c * 2 + 1) * 2 + 0) * 49 + t] = acc10;
    weff[((c * 2 + 1) * 2 + 1) * 49 + t] = acc11;
}

// ---------------------------------------------------------------------------
// Fused per-(band c, batch b, source channel v') pipeline:
//   S = A_c * X * A_c^T  (two chained 128^3 WMMA GEMMs; A symmetric)
//   out[b,0] += conv7x7(S, Weff[c,0,v']);  out[b,1] += conv7x7(S, Weff[c,1,v'])
// A_c and Xt stream into LDS via CDNA5 async global->LDS b128 copies.
// ---------------------------------------------------------------------------
__global__ __launch_bounds__(256) void smooth_conv_kernel(
        const _Float16* __restrict__ Xt,
        const _Float16* __restrict__ Aall,
        const float* __restrict__ weff,
        float* __restrict__ out) {
    const int v = blockIdx.x;
    const int b = blockIdx.y;
    const int c = blockIdx.z;

    extern __shared__ char smem[];
    _Float16* sA  = (_Float16*)smem;                     // 128x128 f16 (32 KB)
    _Float16* sXt = sA + 16384;                          // 128x128 f16 (32 KB), transposed X
    _Float16* sT  = sXt + 16384;                         // 128x128 f16 (32 KB)
    float*    sS  = (float*)(smem + 98304);              // 128x128 f32 (64 KB)
    __shared__ float wV0[49];
    __shared__ float wV1[49];

    const int tid = threadIdx.x;

    // --- Async DMA of A_c and Xt into LDS (16B per lane per op) ---
    {
        const unsigned gss  = __builtin_amdgcn_groupstaticsize();  // dynamic-LDS base
        const unsigned ldsA = gss;
        const unsigned ldsX = gss + 32768u;
        const unsigned long long gA = (unsigned long long)(Aall + (size_t)c * 16384);
        const unsigned long long gX = (unsigned long long)(Xt + ((size_t)(b * 2 + v) << 14));
        for (int i = tid; i < 2048; i += 256) {          // 2048 x 16B = 32 KB each
            unsigned la = ldsA + (unsigned)i * 16u;
            unsigned lx = ldsX + (unsigned)i * 16u;
            unsigned long long ga = gA + (unsigned long long)i * 16ull;
            unsigned long long gx = gX + (unsigned long long)i * 16ull;
            asm volatile("global_load_async_to_lds_b128 %0, %1, off"
                         :: "v"(la), "v"(ga) : "memory");
            asm volatile("global_load_async_to_lds_b128 %0, %1, off"
                         :: "v"(lx), "v"(gx) : "memory");
        }
    }
    if (tid < 49) {
        wV0[tid] = weff[((c * 2 + 0) * 2 + v) * 49 + tid];
        wV1[tid] = weff[((c * 2 + 1) * 2 + v) * 49 + tid];
    }
    asm volatile("s_wait_asynccnt 0" ::: "memory");
    __syncthreads();

    const int wave = tid >> 5;
    const int lane = tid & 31;
    const int ln = lane & 15;           // M (A/C) or N (B/C) lane index
    const int hh = lane >> 4;           // lane-half select
    const int row0 = wave * 16;

    // --- GEMM 1: T = A * X. B-frag from Xt rows (two 16B chunks per lane). ---
    for (int nt = 0; nt < 8; ++nt) {
        const int col0 = nt * 16;
        v8f acc = {};
        #pragma unroll
        for (int ks = 0; ks < 4; ++ks) {
            const int kb = ks * 32;
            // A-frag: two contiguous 8-half chunks: K = kb+8h+{0..7}, kb+16+8h+{0..7}
            const _Float16* ap = sA + (row0 + ln) * 128 + kb + 8 * hh;
            v8h alo = *(const v8h*)(ap);
            v8h ahi = *(const v8h*)(ap + 16);
            v16h af = __builtin_shufflevector(alo, ahi,
                0, 1, 2, 3, 4, 5, 6, 7, 8, 9, 10, 11, 12, 13, 14, 15);
            // B-frag: Xt[col0+ln][kb+8h+{0..7}], Xt[col0+ln][kb+16+8h+{0..7}]
            const _Float16* bp = sXt + (col0 + ln) * 128 + kb + 8 * hh;
            v16h bf = __builtin_shufflevector(*(const v8h*)(bp), *(const v8h*)(bp + 16),
                0, 1, 2, 3, 4, 5, 6, 7, 8, 9, 10, 11, 12, 13, 14, 15);
            acc = __builtin_amdgcn_wmma_f32_16x16x32_f16(
                false, af, false, bf, (short)0, acc, false, false);
        }
        #pragma unroll
        for (int r = 0; r < 8; ++r)      // C/D: element r -> M = r+8h, N = ln
            sT[(row0 + r + 8 * hh) * 128 + col0 + ln] = (_Float16)acc[r];
    }
    // No barrier: wave w produced T rows [16w,16w+16) and consumes only those.

    // --- GEMM 2: S = T * A^T; A symmetric => B'[k,n] = A[n,k] (row read). ---
    for (int nt = 0; nt < 8; ++nt) {
        const int col0 = nt * 16;
        v8f acc = {};
        #pragma unroll
        for (int ks = 0; ks < 4; ++ks) {
            const int kb = ks * 32;
            const _Float16* tp = sT + (row0 + ln) * 128 + kb + 8 * hh;
            v8h tlo = *(const v8h*)(tp);
            v8h thi = *(const v8h*)(tp + 16);
            v16h af = __builtin_shufflevector(tlo, thi,
                0, 1, 2, 3, 4, 5, 6, 7, 8, 9, 10, 11, 12, 13, 14, 15);
            const _Float16* bp = sA + (col0 + ln) * 128 + kb + 8 * hh;
            v16h bf = __builtin_shufflevector(*(const v8h*)(bp), *(const v8h*)(bp + 16),
                0, 1, 2, 3, 4, 5, 6, 7, 8, 9, 10, 11, 12, 13, 14, 15);
            acc = __builtin_amdgcn_wmma_f32_16x16x32_f16(
                false, af, false, bf, (short)0, acc, false, false);
        }
        #pragma unroll
        for (int r = 0; r < 8; ++r)
            sS[(row0 + r + 8 * hh) * 128 + col0 + ln] = acc[r];
    }
    __syncthreads();

    // --- 7x7 zero-padded conv, register-blocked over 1x8 pixel strips ---
    float* out0 = out + ((size_t)(b * 2 + 0) << 14);
    float* out1 = out + ((size_t)(b * 2 + 1) << 14);
    for (int it = 0; it < 8; ++it) {
        const int s  = tid + (it << 8);      // strip id, 2048 strips
        const int y  = s >> 4;
        const int x0 = (s & 15) * 8;
        float acc0[8] = {}, acc1[8] = {};
        #pragma unroll
        for (int ky = 0; ky < 7; ++ky) {
            const int yy = y + ky - 3;
            if ((unsigned)yy > 127u) continue;
            // Window covers x in [x0-4, x0+12): 4 aligned float4 LDS loads.
            const float4* rp = (const float4*)(sS + yy * 128 + x0 - 4);
            float rb[16];
            #pragma unroll
            for (int q = 0; q < 4; ++q) {
                float4 f = rp[q];
                rb[q * 4 + 0] = f.x; rb[q * 4 + 1] = f.y;
                rb[q * 4 + 2] = f.z; rb[q * 4 + 3] = f.w;
            }
            if (x0 == 0)   { rb[0] = rb[1] = rb[2] = rb[3] = 0.0f; }
            if (x0 == 120) { rb[12] = rb[13] = rb[14] = rb[15] = 0.0f; }
            #pragma unroll
            for (int kx = 0; kx < 7; ++kx) {
                const float w0 = wV0[ky * 7 + kx];
                const float w1 = wV1[ky * 7 + kx];
                #pragma unroll
                for (int px = 0; px < 8; ++px) {
                    const float val = rb[px + kx + 1];   // i = (x0+px+kx-3)-(x0-4)
                    acc0[px] = fmaf(w0, val, acc0[px]);
                    acc1[px] = fmaf(w1, val, acc1[px]);
                }
            }
        }
        #pragma unroll
        for (int px = 0; px < 8; ++px) {
            atomicAdd(out0 + y * 128 + x0 + px, acc0[px]);
            atomicAdd(out1 + y * 128 + x0 + px, acc1[px]);
        }
    }
}

// ---------------------------------------------------------------------------
// Host entry. Workspace: [0,512K) A_c (f16); [512K,+12.5K) Weff (f32);
// [768K, 768K+2M) Xt (f16, transposed images).
// ---------------------------------------------------------------------------
extern "C" void kernel_launch(void* const* d_in, const int* in_sizes, int n_in,
                              void* d_out, int out_size, void* d_ws, size_t ws_size,
                              hipStream_t stream) {
    const float* inp     = (const float*)d_in[0];   // [32,2,128,128]
    const float* kernels = (const float*)d_in[1];   // [16,4,1,1,8,7,7]
    const float* rw      = (const float*)d_in[2];   // [16,8,4]
    float* out = (float*)d_out;                     // [32,2,128,128]

    _Float16* Ahalf = (_Float16*)d_ws;
    float*    weff  = (float*)((char*)d_ws + 524288);
    _Float16* Xt    = (_Float16*)((char*)d_ws + 786432);

    zero_kernel<<<4096, 256, 0, stream>>>(out, 32 * 2 * 128 * 128);
    bandmat_kernel<<<16, 256, 0, stream>>>(Ahalf);
    xprep_kernel<<<64, 256, 0, stream>>>(inp, Xt);
    weff_kernel<<<16, 64, 0, stream>>>(kernels, rw, weff);

    dim3 grid(2, 32, 16);   // (v', b, c)
    smooth_conv_kernel<<<grid, 256, 163840, stream>>>(Xt, Ahalf, weff, out);
}
